// SimpleTransformerCPU_4449586119385
// MI455X (gfx1250) — matrix operand order repaired and checked by reading, not compiled, not run on hardware
//
#include <hip/hip_runtime.h>
#include <hip/hip_bf16.h>
#include <cstdint>

// ---------------------------------------------------------------------------
// SimpleTransformer forward on MI455X (gfx1250, wave32, WMMA).
//   x[4,2048,1024] f32 ; W_q/W_k/W_v/fc_out [1024,1024] f32 -> out f32.
// Pipeline:
//   1) x -> bf16 ; weights -> bf16 transposed [N][K]
//   2) QKV GEMMs via v_wmma_f32_16x16x32_bf16, async-to-LDS double-buffered
//   3) flash attention per (b,h) head: WMMA for QK^T and P*V, online softmax
//   4) final GEMM -> f32 d_out
// ---------------------------------------------------------------------------

typedef __bf16 bf16_t;
typedef __attribute__((ext_vector_type(16))) __bf16 v16bf;
typedef __attribute__((ext_vector_type(8)))  __bf16 v8bf;
typedef __attribute__((ext_vector_type(8)))  float  v8f;
typedef __attribute__((ext_vector_type(4)))  int    v4i;

#define D_MODEL  1024
#define HEADS    16
#define HEAD_DIM 64
#define BATCH    4
#define SEQ      2048
#define M_TOTAL  (BATCH * SEQ)   // 8192 tokens

union Frag16 { v16bf v; v8bf h[2]; };

static __device__ __forceinline__ v8f wmma_bf16(v16bf a, v16bf b, v8f c) {
  // D = A(16x32 bf16) * B(32x16 bf16) + C(16x16 f32)
  return __builtin_amdgcn_wmma_f32_16x16x32_bf16(
      /*neg_a=*/false, a, /*neg_b=*/false, b,
      /*c_mod=*/(short)0, c, /*reuse_a=*/false, /*reuse_b=*/false);
}

// ---- CDNA5 async global->LDS copy (ASYNCcnt path).  The builtin's prototype
// (from the round-2 diagnostic) is (v4i addrspace(1)*, v4i addrspace(3)*,
// imm offset, imm cpol).  Falls back to load->ds_store staging if absent.
#if __has_builtin(__builtin_amdgcn_global_load_async_to_lds_b128)
#define HAVE_ASYNC_COPY 1
typedef __attribute__((address_space(1))) v4i* as1_v4ip;
typedef __attribute__((address_space(3))) v4i* as3_v4ip;
static __device__ __forceinline__ void async_copy16(const bf16_t* g, bf16_t* l) {
  // 16 bytes, global -> LDS, no VGPR data movement.
  __builtin_amdgcn_global_load_async_to_lds_b128(
      (as1_v4ip)(uintptr_t)g,
      (as3_v4ip)(uint32_t)(uintptr_t)l,
      /*offset=*/0, /*cpol=*/0);
}
#if __has_builtin(__builtin_amdgcn_s_wait_asynccnt)
#define WAIT_ASYNC(n) __builtin_amdgcn_s_wait_asynccnt(n)
#else
#define WAIT_ASYNC(n) asm volatile("s_wait_asynccnt %0" ::"i"(n) : "memory")
#endif
#else
#define HAVE_ASYNC_COPY 0
#define WAIT_ASYNC(n) do {} while (0)
static __device__ __forceinline__ void async_copy16(const bf16_t* g, bf16_t* l) {
  *(v8bf*)l = *(const v8bf*)g;   // synchronous fallback
}
#endif

// ---------------------------------------------------------------- converts --
__global__ __launch_bounds__(256) void k_f32_to_bf16(
    const float* __restrict__ in, bf16_t* __restrict__ out, int n) {
  for (int i = blockIdx.x * blockDim.x + threadIdx.x; i < n;
       i += gridDim.x * blockDim.x)
    out[i] = (bf16_t)in[i];
}

// Wt[n][k] = W[k][n], f32 -> bf16, 32x32 LDS-tiled transpose.
__global__ __launch_bounds__(256) void k_transpose_to_bf16(
    const float* __restrict__ W, bf16_t* __restrict__ Wt, int dim) {
  __shared__ float tile[32][33];
  const int bx = blockIdx.x * 32, by = blockIdx.y * 32;
  const int tx = threadIdx.x & 31;
  const int ty4 = (threadIdx.x >> 5) * 4;
#pragma unroll
  for (int i = 0; i < 4; ++i)
    tile[ty4 + i][tx] = W[(size_t)(by + ty4 + i) * dim + bx + tx];
  __syncthreads();
#pragma unroll
  for (int i = 0; i < 4; ++i)
    Wt[(size_t)(bx + ty4 + i) * dim + by + tx] = (bf16_t)tile[tx][ty4 + i];
}

// -------------------------------------------------------------------- GEMM --
// C[M,N] = A[M,K] * B[K,N], with B given transposed Bt[N][K] (both bf16).
// Block: 128x128 tile, 256 threads = 8 waves, wave = 64x32 sub-tile
// (4 M-tiles x 2 N-tiles of 16x16), k-step 32.  LDS double-buffered: tile
// kt+32 streams in on the async engine while WMMA consumes tile kt.
// mode 0: write f32 row-major to Cf.
// mode 1: write bf16 head-split [b*HEADS+h][s][HEAD_DIM] to Chs (QKV path).
__global__ __launch_bounds__(256) void k_gemm_bf16(
    const bf16_t* __restrict__ A, const bf16_t* __restrict__ Bt,
    float* __restrict__ Cf, bf16_t* __restrict__ Chs,
    int M, int N, int Kd, int mode) {
  __shared__ bf16_t As[2][128][40];   // +8 bf16 pad: 16B-aligned, conflict-free
  __shared__ bf16_t Bs[2][128][40];

  const int tid  = threadIdx.x;
  const int lane = tid & 31;
  const int wave = tid >> 5;
  const int m0 = blockIdx.x * 128;
  const int n0 = blockIdx.y * 128;
  const int wm = (wave >> 2) * 64;   // wave row offset   (0 / 64)
  const int wn = (wave & 3) * 32;    // wave col offset   (0/32/64/96)

  const v8f zero8 = {0.f,0.f,0.f,0.f,0.f,0.f,0.f,0.f};
  v8f acc[4][2];
#pragma unroll
  for (int mt = 0; mt < 4; ++mt)
#pragma unroll
    for (int nt = 0; nt < 2; ++nt) acc[mt][nt] = zero8;

  const int ldr  = tid >> 1;           // staging row 0..127
  const int lcol = (tid & 1) * 16;     // staging col 0 / 16 (16 bf16 each)

  // stage one 128x32 A tile + 128x32 B tile into buffer `buf`
  // (4 async b128 ops per thread -> ASYNCcnt == 4 while in flight)
  auto stage = [&](int buf, int kt) {
    async_copy16(&A [(size_t)(m0 + ldr) * Kd + kt + lcol],     &As[buf][ldr][lcol]);
    async_copy16(&A [(size_t)(m0 + ldr) * Kd + kt + lcol + 8], &As[buf][ldr][lcol + 8]);
    async_copy16(&Bt[(size_t)(n0 + ldr) * Kd + kt + lcol],     &Bs[buf][ldr][lcol]);
    async_copy16(&Bt[(size_t)(n0 + ldr) * Kd + kt + lcol + 8], &Bs[buf][ldr][lcol + 8]);
  };

  stage(0, 0);                         // prologue fill

  for (int kt = 0; kt < Kd; kt += 32) {
    const int buf = (kt >> 5) & 1;
    if (kt + 32 < Kd) {
      stage(buf ^ 1, kt + 32);         // async prefetch of next tile
      __builtin_prefetch(&A [(size_t)(m0 + ldr) * Kd + kt + 64 + lcol], 0, 3);
      __builtin_prefetch(&Bt[(size_t)(n0 + ldr) * Kd + kt + 64 + lcol], 0, 3);
      WAIT_ASYNC(4);                   // current tile done; next may stay in flight
    } else {
      WAIT_ASYNC(0);
    }
    __syncthreads();                   // everyone's stage of `buf` visible

    // A fragments: lane row = lane&15, K = (lane>>4)*16 + {0..15}
    Frag16 af[4];
    const int kb = (lane >> 4) * 16;
#pragma unroll
    for (int mt = 0; mt < 4; ++mt) {
      const int row = wm + mt * 16 + (lane & 15);
      af[mt].h[0] = *(const v8bf*)&As[buf][row][kb];
      af[mt].h[1] = *(const v8bf*)&As[buf][row][kb + 8];
    }
    // B fragments: lane col = lane&15, K = (lane>>4)*16 + {0..15}
    Frag16 bf[2];
    const int kb2 = (lane >> 4) * 16;
#pragma unroll
    for (int nt = 0; nt < 2; ++nt) {
      const int col = wn + nt * 16 + (lane & 15);
      bf[nt].h[0] = *(const v8bf*)&Bs[buf][col][kb2];
      bf[nt].h[1] = *(const v8bf*)&Bs[buf][col][kb2 + 8];
    }
#pragma unroll
    for (int mt = 0; mt < 4; ++mt)
#pragma unroll
      for (int nt = 0; nt < 2; ++nt)
        acc[mt][nt] = wmma_bf16(af[mt].v, bf[nt].v, acc[mt][nt]);
    __syncthreads();                   // done reading `buf` before it is restaged
  }

  // epilogue: C-layout row = (lane>>4)*8 + r, col = lane&15
#pragma unroll
  for (int mt = 0; mt < 4; ++mt)
#pragma unroll
    for (int nt = 0; nt < 2; ++nt)
#pragma unroll
      for (int r = 0; r < 8; ++r) {
        const int row = m0 + wm + mt * 16 + (lane >> 4) * 8 + r;
        const int col = n0 + wn + nt * 16 + (lane & 15);
        const float v = acc[mt][nt][r];
        if (mode == 0) {
          Cf[(size_t)row * N + col] = v;
        } else {
          const int b = row >> 11, s = row & (SEQ - 1);   // SEQ = 2048
          const int h = col >> 6, d = col & (HEAD_DIM - 1);
          Chs[(((size_t)(b * HEADS + h)) * SEQ + s) * HEAD_DIM + d] = (bf16_t)v;
        }
      }
}

// --------------------------------------------------------- flash attention --
// grid.x = b*h (64), grid.y = query tiles of 128.  8 waves; each wave owns a
// 16-row query strip, streams 64-key tiles: S = Q K^T (WMMA), online softmax
// (half-wave shfl reductions match the C-matrix row layout), P -> LDS bf16,
// O += P V (WMMA).  Scores never touch HBM.  Q/K tiles stream in on the
// async-to-LDS engine; V needs an in-LDS transpose so it stays manual.
__global__ __launch_bounds__(256) void k_attention(
    const bf16_t* __restrict__ Qh, const bf16_t* __restrict__ Kh,
    const bf16_t* __restrict__ Vh, bf16_t* __restrict__ concat) {
  __shared__ bf16_t Qs[128][72];      // query tile      (stride 72 -> 16B aligned)
  __shared__ bf16_t Ks[64][72];       // key tile, already Bt-form [key][d]
  __shared__ bf16_t Vt[64][72];       // V transposed    [d][key]
  __shared__ bf16_t Ps[8][16][72];    // per-wave P strip for A-operand reload

  const int bh   = blockIdx.x;              // 0..63  (b*16 + h)
  const int q0   = blockIdx.y * 128;
  const int tid  = threadIdx.x;
  const int lane = tid & 31;
  const int wave = tid >> 5;

  const bf16_t* Qg = Qh + (size_t)bh * SEQ * HEAD_DIM;
  const bf16_t* Kg = Kh + (size_t)bh * SEQ * HEAD_DIM;
  const bf16_t* Vg = Vh + (size_t)bh * SEQ * HEAD_DIM;

  { // stage Q tile 128x64 (each thread 32 bf16 = 4 async b128 ops)
    const int row = tid >> 1, c0 = (tid & 1) * 32;
#pragma unroll
    for (int i = 0; i < 32; i += 8)
      async_copy16(&Qg[(size_t)(q0 + row) * HEAD_DIM + c0 + i], &Qs[row][c0 + i]);
    WAIT_ASYNC(0);
  }
  __syncthreads();

  // per-wave Q A-fragments, 2 k-steps over d = 0..63
  Frag16 aq[2];
  {
    const int qrow = wave * 16 + (lane & 15);
    const int kb = (lane >> 4) * 16;
#pragma unroll
    for (int ks = 0; ks < 2; ++ks) {
      aq[ks].h[0] = *(const v8bf*)&Qs[qrow][ks * 32 + kb];
      aq[ks].h[1] = *(const v8bf*)&Qs[qrow][ks * 32 + kb + 8];
    }
  }

  const v8f zero8 = {0.f,0.f,0.f,0.f,0.f,0.f,0.f,0.f};
  v8f o[4];                           // 16 rows x 64 cols of O accumulator
#pragma unroll
  for (int nt = 0; nt < 4; ++nt) o[nt] = zero8;
  float mrow[8], srow[8];             // running max / sum per owned row
#pragma unroll
  for (int r = 0; r < 8; ++r) { mrow[r] = -1e30f; srow[r] = 0.f; }

  const float scale = 0.125f;         // 1/sqrt(HEAD_DIM)
  const int kb2 = (lane >> 4) * 16;

  for (int kt = 0; kt < SEQ; kt += 64) {
    __syncthreads();                  // protect Ks/Vt from previous iteration
    { // stage K tile [key][d] directly (async)
      const int r = tid >> 2, c = (tid & 3) * 16;
      async_copy16(&Kg[(size_t)(kt + r) * HEAD_DIM + c],     &Ks[r][c]);
      async_copy16(&Kg[(size_t)(kt + r) * HEAD_DIM + c + 8], &Ks[r][c + 8]);
    }
    { // stage V transposed [d][key]
      const int r = tid >> 2, c0 = (tid & 3) * 16;
#pragma unroll
      for (int j = 0; j < 16; ++j)
        Vt[c0 + j][r] = Vg[(size_t)(kt + r) * HEAD_DIM + c0 + j];
    }
    WAIT_ASYNC(0);
    __syncthreads();

    // S = Q * K^T  (16 x 64 strip per wave)
    v8f s[4];
#pragma unroll
    for (int nt = 0; nt < 4; ++nt) {
      const int col = nt * 16 + (lane & 15);
      Frag16 bk0, bk1;
      bk0.h[0] = *(const v8bf*)&Ks[col][kb2];
      bk0.h[1] = *(const v8bf*)&Ks[col][kb2 + 8];
      bk1.h[0] = *(const v8bf*)&Ks[col][32 + kb2];
      bk1.h[1] = *(const v8bf*)&Ks[col][32 + kb2 + 8];
      v8f z = zero8;
      z = wmma_bf16(aq[0].v, bk0.v, z);
      z = wmma_bf16(aq[1].v, bk1.v, z);
      s[nt] = z;
    }

    // online softmax per owned row (row spans 16 lanes of one half-wave)
#pragma unroll
    for (int r = 0; r < 8; ++r) {
      float sv[4], mx = -1e30f;
#pragma unroll
      for (int nt = 0; nt < 4; ++nt) { sv[nt] = s[nt][r] * scale; mx = fmaxf(mx, sv[nt]); }
#pragma unroll
      for (int off = 1; off < 16; off <<= 1)
        mx = fmaxf(mx, __shfl_xor(mx, off, 16));
      const float mnew = fmaxf(mrow[r], mx);
      const float corr = __expf(mrow[r] - mnew);
      float p[4], psum = 0.f;
#pragma unroll
      for (int nt = 0; nt < 4; ++nt) { p[nt] = __expf(sv[nt] - mnew); psum += p[nt]; }
#pragma unroll
      for (int off = 1; off < 16; off <<= 1)
        psum += __shfl_xor(psum, off, 16);
      srow[r] = srow[r] * corr + psum;
      mrow[r] = mnew;
#pragma unroll
      for (int nt = 0; nt < 4; ++nt) o[nt][r] *= corr;
      const int prow = (lane >> 4) * 8 + r;
#pragma unroll
      for (int nt = 0; nt < 4; ++nt)
        Ps[wave][prow][nt * 16 + (lane & 15)] = (bf16_t)p[nt];
    }

    // O += P * V : reload P in A-operand layout (same-wave LDS, dscnt-ordered)
    Frag16 ap[2];
    {
      const int prow = lane & 15, kbp = (lane >> 4) * 16;
#pragma unroll
      for (int ks = 0; ks < 2; ++ks) {
        ap[ks].h[0] = *(const v8bf*)&Ps[wave][prow][ks * 32 + kbp];
        ap[ks].h[1] = *(const v8bf*)&Ps[wave][prow][ks * 32 + kbp + 8];
      }
    }
#pragma unroll
    for (int nt = 0; nt < 4; ++nt) {
      const int col = nt * 16 + (lane & 15);
      Frag16 bv0, bv1;
      bv0.h[0] = *(const v8bf*)&Vt[col][kb2];
      bv0.h[1] = *(const v8bf*)&Vt[col][kb2 + 8];
      bv1.h[0] = *(const v8bf*)&Vt[col][32 + kb2];
      bv1.h[1] = *(const v8bf*)&Vt[col][32 + kb2 + 8];
      o[nt] = wmma_bf16(ap[0].v, bv0.v, o[nt]);
      o[nt] = wmma_bf16(ap[1].v, bv1.v, o[nt]);
    }
  }

  // normalize and write concat[b][s][h*64 + d] as bf16
  const int b = bh >> 4, h = bh & 15;
#pragma unroll
  for (int nt = 0; nt < 4; ++nt)
#pragma unroll
    for (int r = 0; r < 8; ++r) {
      const int q = q0 + wave * 16 + (lane >> 4) * 8 + r;
      const int d = nt * 16 + (lane & 15);
      const float val = o[nt][r] / srow[r];
      concat[((size_t)(b * SEQ + q)) * D_MODEL + h * HEAD_DIM + d] = (bf16_t)val;
    }
}

// ------------------------------------------------------------------ launch --
extern "C" void kernel_launch(void* const* d_in, const int* in_sizes, int n_in,
                              void* d_out, int out_size, void* d_ws, size_t ws_size,
                              hipStream_t stream) {
  const float* x   = (const float*)d_in[0];
  const float* Wq  = (const float*)d_in[1];
  const float* Wk  = (const float*)d_in[2];
  const float* Wv  = (const float*)d_in[3];
  const float* Wo  = (const float*)d_in[4];
  float* out = (float*)d_out;

  // workspace carve-up (bf16), ~92 MB total
  char* ws = (char*)d_ws;
  size_t ofs = 0;
  auto carve = [&](size_t bytes) { char* p = ws + ofs; ofs += bytes; return p; };
  const size_t XB   = (size_t)M_TOTAL * D_MODEL * sizeof(bf16_t);   // 16 MB
  const size_t WB   = (size_t)D_MODEL * D_MODEL * sizeof(bf16_t);   // 2 MB
  bf16_t* xb     = (bf16_t*)carve(XB);
  bf16_t* Wq_t   = (bf16_t*)carve(WB);
  bf16_t* Wk_t   = (bf16_t*)carve(WB);
  bf16_t* Wv_t   = (bf16_t*)carve(WB);
  bf16_t* Wo_t   = (bf16_t*)carve(WB);
  bf16_t* Qhs    = (bf16_t*)carve(XB);
  bf16_t* Khs    = (bf16_t*)carve(XB);
  bf16_t* Vhs    = (bf16_t*)carve(XB);
  bf16_t* concat = (bf16_t*)carve(XB);

  // 1) downconvert / transpose
  k_f32_to_bf16<<<4096, 256, 0, stream>>>(x, xb, M_TOTAL * D_MODEL);
  dim3 tg(D_MODEL / 32, D_MODEL / 32);
  k_transpose_to_bf16<<<tg, 256, 0, stream>>>(Wq, Wq_t, D_MODEL);
  k_transpose_to_bf16<<<tg, 256, 0, stream>>>(Wk, Wk_t, D_MODEL);
  k_transpose_to_bf16<<<tg, 256, 0, stream>>>(Wv, Wv_t, D_MODEL);
  k_transpose_to_bf16<<<tg, 256, 0, stream>>>(Wo, Wo_t, D_MODEL);

  // 2) QKV projections (head-split bf16 outputs)
  dim3 gg(M_TOTAL / 128, D_MODEL / 128);
  k_gemm_bf16<<<gg, 256, 0, stream>>>(xb, Wq_t, nullptr, Qhs, M_TOTAL, D_MODEL, D_MODEL, 1);
  k_gemm_bf16<<<gg, 256, 0, stream>>>(xb, Wk_t, nullptr, Khs, M_TOTAL, D_MODEL, D_MODEL, 1);
  k_gemm_bf16<<<gg, 256, 0, stream>>>(xb, Wv_t, nullptr, Vhs, M_TOTAL, D_MODEL, D_MODEL, 1);

  // 3) flash attention: 64 heads x 16 query tiles
  k_attention<<<dim3(BATCH * HEADS, SEQ / 128), 256, 0, stream>>>(Qhs, Khs, Vhs, concat);

  // 4) output projection -> f32
  k_gemm_bf16<<<gg, 256, 0, stream>>>(concat, Wo_t, out, nullptr, M_TOTAL, D_MODEL, D_MODEL, 0);
}
